// Encoder_33036888441301
// MI455X (gfx1250) — matrix operand order repaired and checked by reading, not compiled, not run on hardware
//
#include <hip/hip_runtime.h>
#include <hip/hip_bf16.h>

// ---------------------------------------------------------------------------
// Geometric-attention encoder layer for MI455X (gfx1250, wave32, WMMA).
// All matmuls use v_wmma_f32_16x16x32_f16; attention is flash-style so the
// B*H*N*N score tensor (268MB) is never materialized (keeps us on the
// ~150MB / 23.3TB/s memory roofline, all resident in the 192MB L2).
// K-tile staging in attention uses the CDNA5 Tensor Data Mover
// (tensor_load_to_lds + s_wait_tensorcnt) with double buffering; GEMMs use
// double-buffered LDS staging + global_prefetch_b8 lookahead and
// compile-time epilogue flags (no runtime branches in the hot path).
// ---------------------------------------------------------------------------

typedef _Float16 h16;
typedef _Float16 h8v  __attribute__((ext_vector_type(8)));
typedef _Float16 v16h __attribute__((ext_vector_type(16)));
typedef float    v8f  __attribute__((ext_vector_type(8)));
typedef unsigned int u32x4 __attribute__((ext_vector_type(4)));
typedef int          i32x4 __attribute__((ext_vector_type(4)));
typedef int          i32x8 __attribute__((ext_vector_type(8)));

#define B_     2
#define N_     2048
#define DM     512
#define HEADS_ 8
#define DK     64
#define DH     2048
#define ROWS   (B_ * N_)   // 4096

__device__ inline v8f zero8() {
  v8f z;
#pragma unroll
  for (int i = 0; i < 8; ++i) z[i] = 0.0f;
  return z;
}

__device__ inline v16h frag_from(const h16* lo, const h16* hi) {
  h8v l0 = *(const h8v*)lo, h0 = *(const h8v*)hi;
  v16h f;
#pragma unroll
  for (int i = 0; i < 8; ++i) { f[i] = l0[i]; f[i + 8] = h0[i]; }
  return f;
}

// ---- Tensor Data Mover: 2D f16 tile, global -> LDS (CDNA5 D# descriptor) ---
// lds_addr: workgroup-relative LDS byte address (low 32 bits of flat ptr).
// tile_d0 elems/row, tile_d1 rows; stride_elems = global row stride.
// Optional LDS padding: pad_amount DWORD-encoded, every (1<<pad_interval)*8B.
__device__ inline void tdm_load_tile_f16(unsigned lds_addr, const void* gptr,
                                         unsigned tile_d0, unsigned tile_d1,
                                         unsigned stride_elems,
                                         unsigned pad_en, unsigned pad_interval,
                                         unsigned pad_amount) {
  unsigned long long ga = (unsigned long long)(uintptr_t)gptr;
  u32x4 g0;
  g0[0] = 1u;                                               // count=1 (valid user D#)
  g0[1] = lds_addr;                                         // lds_addr[31:0]
  g0[2] = (unsigned)(ga & 0xFFFFFFFFu);                     // global_addr[31:0]
  g0[3] = (unsigned)((ga >> 32) & 0x01FFFFFFu) | (2u << 30);// global_addr[56:32] | type=2
  i32x8 g1;
  g1[0] = (int)((1u << 16) | (pad_en << 20) |               // data_size=1 (2B), pad cfg
                (pad_interval << 22) | (pad_amount << 25));
  g1[1] = (int)((tile_d0 & 0xFFFFu) << 16);                 // tensor_dim0 = tile_d0 (lo16)
  g1[2] = (int)(((tile_d0 >> 16) & 0xFFFFu) |               // tensor_dim0 hi16
                ((tile_d1 & 0xFFFFu) << 16));               // tensor_dim1 = tile_d1 (lo16)
  g1[3] = (int)(((tile_d1 >> 16) & 0xFFFFu) |               // tensor_dim1 hi16
                ((tile_d0 & 0xFFFFu) << 16));               // tile_dim0
  g1[4] = (int)(tile_d1 & 0xFFFFu);                         // tile_dim1 (tile_dim2=0)
  g1[5] = (int)stride_elems;                                // tensor_dim0_stride lo32
  g1[6] = 0;                                                // stride0 hi / stride1 lo
  g1[7] = 0;
  i32x4 z4 = {0, 0, 0, 0};
#if __clang_major__ >= 23
  i32x8 z8 = {0, 0, 0, 0, 0, 0, 0, 0};
  __builtin_amdgcn_tensor_load_to_lds(g0, g1, z4, z4, z8, 0);
#else
  __builtin_amdgcn_tensor_load_to_lds(g0, g1, z4, z4, 0);
#endif
}

// ------------------------- converters --------------------------------------

__global__ void k_cvt_plain(const float* __restrict__ src, h16* __restrict__ dst, int n) {
  int i = blockIdx.x * 256 + threadIdx.x;
  if (i < n) dst[i] = (h16)src[i];
}

// q/k/v_proj [H][DM][DK] -> W[d][h*64+kk]  (so projection is one 512x512 GEMM)
__global__ void k_cvt_headw(const float* __restrict__ src, h16* __restrict__ dst) {
  int i = blockIdx.x * 256 + threadIdx.x;           // 512*512 elements
  int h = i >> 15, d = (i >> 6) & 511, kk = i & 63;
  dst[(d << 9) + (h << 6) + kk] = (h16)src[i];
}

// out_w [j][c] -> Wo[c][j]  (so out @ out_w.T is plain row-major A x B)
__global__ void k_cvt_transpose(const float* __restrict__ src, h16* __restrict__ dst,
                                int R, int C) {
  int i = blockIdx.x * 256 + threadIdx.x;
  if (i < R * C) { int r = i / C, c = i % C; dst[(size_t)c * R + r] = (h16)src[i]; }
}

// ------------------------- generic WMMA GEMM -------------------------------
// C[M,N] = A[M,K](f16,row) x B[K,N](f16,row); epilogue behavior is template-
// resolved (bias / ReLU / f32 store / f16 store) so the hot loop + epilogue
// are branch-free. WG = 128 threads (4 waves); WG tile 64x64, wave tile
// 16x64, K-chunk 32, double-buffered LDS + prefetch lookahead.

template <bool HAS_BIAS, bool RELU, bool WF32, bool WF16>
__global__ __launch_bounds__(128) void k_gemm(
    const h16* __restrict__ A, const h16* __restrict__ Bm,
    const float* __restrict__ bias, float* __restrict__ Cf, h16* __restrict__ Ch,
    int M, int N, int K) {
  __shared__ alignas(16) h16 As[2][64][40];   // [buf][m][k] (pad 8 -> 80B stride)
  __shared__ alignas(16) h16 BT[2][64][40];   // [buf][n][k] transposed B chunk

  const int t = threadIdx.x;
  const int lane = t & 31, wv = t >> 5;
  const int nbase = blockIdx.x * 64, mbase = blockIdx.y * 64;
  const int m      = lane & 15;
  const int kb     = (lane >> 4) * 8;     // A-fragment K base
  const int kstart = (lane >> 4) * 16;    // unused; B fragments also use kb
  const int ar = t >> 1, ahalf = t & 1;   // A staging coords (64 rows x 2 halves)
  const int bk = t >> 2, bseg = t & 3;    // B staging coords (32 k x 4 segs)

  v8f acc[4];
#pragma unroll
  for (int c = 0; c < 4; ++c) acc[c] = zero8();

  auto stage = [&](int buf, int kc) {
    const h8v* src = (const h8v*)(A + (size_t)(mbase + ar) * K + kc + ahalf * 16);
    h8v* dst = (h8v*)(&As[buf][ar][ahalf * 16]);
    dst[0] = src[0]; dst[1] = src[1];
    const h16* bs = Bm + (size_t)(kc + bk) * N + nbase + bseg * 16;
#pragma unroll
    for (int j = 0; j < 16; ++j) BT[buf][bseg * 16 + j][bk] = bs[j];
  };

  const int nch = K / 32;
  stage(0, 0);
  for (int i = 0; i < nch; ++i) {
    const int buf = i & 1;
    __syncthreads();                       // staging of `buf` complete
    if (i + 1 < nch) stage(buf ^ 1, (i + 1) * 32);
    if (i + 2 < nch) {                     // L2/WGP$ lookahead: global_prefetch_b8
      __builtin_prefetch(A + (size_t)(mbase + ar) * K + (i + 2) * 32 + ahalf * 16, 0, 3);
      __builtin_prefetch(Bm + (size_t)((i + 2) * 32 + bk) * N + nbase + bseg * 16, 0, 3);
    }

    const v16h af = frag_from(&As[buf][wv * 16 + m][kb], &As[buf][wv * 16 + m][16 + kb]);
    v16h bf[4];
#pragma unroll
    for (int c = 0; c < 4; ++c)
      bf[c] = frag_from(&BT[buf][c * 16 + m][kb], &BT[buf][c * 16 + m][16 + kb]);
#pragma unroll
    for (int c = 0; c < 4; ++c)
      acc[c] = __builtin_amdgcn_wmma_f32_16x16x32_f16(false, af, false, bf[c],
                                                      (short)0, acc[c], false, false);
  }

  const int rbase = (lane >> 4) * 8;
#pragma unroll
  for (int c = 0; c < 4; ++c) {
    int col = nbase + c * 16 + m;
    float bv = 0.0f;
    if constexpr (HAS_BIAS) bv = bias[col];
#pragma unroll
    for (int r = 0; r < 8; ++r) {
      int row = mbase + wv * 16 + rbase + r;
      float v = acc[c][r] + bv;
      if constexpr (RELU) v = v > 0.0f ? v : 0.0f;
      if constexpr (WF32) Cf[(size_t)row * N + col] = v;
      if constexpr (WF16) Ch[(size_t)row * N + col] = (h16)v;
    }
  }
}

// ------------------------- flash geometric attention -----------------------
// grid: (N/64, HEADS, B); WG = 128 threads (4 waves), 16 query rows per wave.
// Streams 32-key tiles; the K tile is DMA'd by the Tensor Data Mover into a
// double-buffered LDS tile (with DMA-applied row padding), overlapped with
// compute; V is staged transposed by the waves. Fuses beta*rbf scaling +
// rbf>=1e-3 masking into the online softmax. Output f16, (b, n, dk, h).

__global__ __launch_bounds__(128) void k_attn(
    const h16* __restrict__ Qh, const h16* __restrict__ Kh, const h16* __restrict__ Vh,
    const float* __restrict__ coords,
    const float* __restrict__ spread_w, const float* __restrict__ beta_w,
    h16* __restrict__ attnA) {
  __shared__ alignas(16) h16   Kt[2][32][72];     // [buf][key][dk] (TDM destination)
  __shared__ alignas(16) h16   VT[64][40];        // [dk][key] (transposed)
  __shared__ alignas(16) float Kc[32][4];         // key coords
  __shared__ alignas(16) h16   Pscr[4][16][40];   // per-wave P transpose scratch

  const int t = threadIdx.x, lane = t & 31, wv = t >> 5;
  const int h = blockIdx.y, b = blockIdx.z;

  const float spread = 2.0f + __expf(spread_w[h]);
  const float beta   = __expf(beta_w[h]);
  const float inv2s2 = 1.0f / (2.0f * spread * spread);
  const float d2max  = 6.90775527898f / inv2s2;   // rbf >= 1e-3  <=>  d2 <= 2*s^2*ln(1000)
  const float rsdk   = 0.125f;                    // 1/sqrt(64)

  const int qbase  = blockIdx.x * 64;
  const int m      = lane & 15;
  const int kb     = (lane >> 4) * 8;
  const int kstart = (lane >> 4) * 16;
  const int rbase  = (lane >> 4) * 8;
  const int qrow0  = qbase + wv * 16;
  const int skey = t >> 2, sseg = t & 3;          // staging coords

  // K-tile rows are 64 f16 = 128B = 32 DWORDs; pad 16B per row in LDS
  // -> stride 72 f16. pad_interval: 1<<4 * 8B = 128B; pad_amount: 3 -> 4 DWORDs.
  auto tdm_k = [&](int buf, int kt) {
    tdm_load_tile_f16((unsigned)(uintptr_t)&Kt[buf][0][0],
                      Kh + (size_t)(b * N_ + kt) * DM + h * DK,
                      /*tile_d0=*/DK, /*tile_d1=*/32, /*stride=*/DM,
                      /*pad_en=*/1, /*pad_interval=*/4, /*pad_amount=*/3);
  };

  // Q fragments (dk chunks 0-31, 32-63), loaded once
  v16h aq[2];
  {
    const h16* qp = Qh + (size_t)(b * N_ + qrow0 + m) * DM + h * DK;
#pragma unroll
    for (int c = 0; c < 2; ++c)
      aq[c] = frag_from(qp + c * 32 + kb, qp + c * 32 + 16 + kb);
  }
  // per-lane query coords for the 8 C-fragment rows this lane owns
  float qx[8], qy[8], qz[8];
#pragma unroll
  for (int r = 0; r < 8; ++r) {
    const float* cp = coords + (size_t)(b * N_ + qrow0 + rbase + r) * 3;
    qx[r] = cp[0]; qy[r] = cp[1]; qz[r] = cp[2];
  }

  float mrow[8], lrow[8];
  v8f acc[4];
#pragma unroll
  for (int r = 0; r < 8; ++r) { mrow[r] = -1e30f; lrow[r] = 0.0f; }
#pragma unroll
  for (int c = 0; c < 4; ++c) acc[c] = zero8();

  if (wv == 0) tdm_k(0, 0);            // prologue DMA for first K tile

  for (int kt = 0; kt < N_; kt += 32) {
    const int buf = (kt >> 5) & 1;
    if (wv == 0) __builtin_amdgcn_s_wait_tensorcnt(0);   // K tile[buf] landed
    __syncthreads();                    // publish K tile; prev compute done
    {   // stage V tile (transposed) by the waves
      const h16* vs = Vh + (size_t)(b * N_ + kt + skey) * DM + h * DK + sseg * 16;
#pragma unroll
      for (int j = 0; j < 16; ++j) VT[sseg * 16 + j][skey] = vs[j];
    }
    if (t < 32) {
      const float* cp = coords + (size_t)(b * N_ + kt + t) * 3;
      Kc[t][0] = cp[0]; Kc[t][1] = cp[1]; Kc[t][2] = cp[2];
    }
    if (wv == 0 && kt + 32 < N_) tdm_k(buf ^ 1, kt + 32);   // async next K tile
    if (kt + 32 < N_) {   // lookahead prefetch of next V tile (global_prefetch_b8)
      __builtin_prefetch(Vh + (size_t)(b * N_ + kt + 32 + skey) * DM + h * DK + sseg * 16, 0, 3);
    }
    __syncthreads();                    // VT/Kc ready

    // ---- scores S = Q K^T for two 16-key halves ----
    v16h bkf[2][2];   // [half][dk-chunk] fragments, loaded up front
#pragma unroll
    for (int half = 0; half < 2; ++half) {
      int lkey = half * 16 + m;
#pragma unroll
      for (int c = 0; c < 2; ++c)
        bkf[half][c] = frag_from(&Kt[buf][lkey][c * 32 + kb],
                                 &Kt[buf][lkey][c * 32 + 16 + kb]);
    }
    float sc[2][8], tmax[8];
#pragma unroll
    for (int r = 0; r < 8; ++r) tmax[r] = -1e30f;

#pragma unroll
    for (int half = 0; half < 2; ++half) {
      v8f s = zero8();
      s = __builtin_amdgcn_wmma_f32_16x16x32_f16(false, aq[0], false, bkf[half][0],
                                                 (short)0, s, false, false);
      s = __builtin_amdgcn_wmma_f32_16x16x32_f16(false, aq[1], false, bkf[half][1],
                                                 (short)0, s, false, false);
      int lkey = half * 16 + m;
      float kx = Kc[lkey][0], ky = Kc[lkey][1], kz = Kc[lkey][2];
#pragma unroll
      for (int r = 0; r < 8; ++r) {
        float dx = qx[r] - kx, dy = qy[r] - ky, dz = qz[r] - kz;
        float d2 = dx * dx + dy * dy + dz * dz;
        float v;
        if (d2 <= d2max) {
          float rbf = __expf(-d2 * inv2s2);
          v = beta * rbf * (s[r] * rsdk);
        } else {
          v = -1e30f;                    // masked out (rbf < 1e-3)
        }
        sc[half][r] = v;
        tmax[r] = fmaxf(tmax[r], v);
      }
    }
    // row-max across the 16 lanes holding this row's columns
#pragma unroll
    for (int r = 0; r < 8; ++r) {
      float v = tmax[r];
      v = fmaxf(v, __shfl_xor(v, 1));
      v = fmaxf(v, __shfl_xor(v, 2));
      v = fmaxf(v, __shfl_xor(v, 4));
      v = fmaxf(v, __shfl_xor(v, 8));
      tmax[r] = v;
    }
    float alpha[8], psum[8];
#pragma unroll
    for (int r = 0; r < 8; ++r) {
      float nm = fmaxf(mrow[r], tmax[r]);
      alpha[r] = __expf(mrow[r] - nm);
      mrow[r] = nm;
      psum[r] = 0.0f;
    }
#pragma unroll
    for (int half = 0; half < 2; ++half) {
#pragma unroll
      for (int r = 0; r < 8; ++r) {
        float p = (sc[half][r] > -1e29f) ? __expf(sc[half][r] - mrow[r]) : 0.0f;
        psum[r] += p;
        Pscr[wv][rbase + r][half * 16 + m] = (h16)p;   // C-layout -> LDS
      }
    }
#pragma unroll
    for (int r = 0; r < 8; ++r) {
      float v = psum[r];
      v += __shfl_xor(v, 1);
      v += __shfl_xor(v, 2);
      v += __shfl_xor(v, 4);
      v += __shfl_xor(v, 8);
      lrow[r] = lrow[r] * alpha[r] + v;
#pragma unroll
      for (int c = 0; c < 4; ++c) acc[c][r] *= alpha[r];
    }
    __syncthreads();   // Pscr visible across lanes; VT reads below are safe

    const v16h pf = frag_from(&Pscr[wv][m][kb], &Pscr[wv][m][16 + kb]);
    v16h bvf[4];
#pragma unroll
    for (int c = 0; c < 4; ++c)
      bvf[c] = frag_from(&VT[c * 16 + m][kb], &VT[c * 16 + m][16 + kb]);
#pragma unroll
    for (int c = 0; c < 4; ++c)
      acc[c] = __builtin_amdgcn_wmma_f32_16x16x32_f16(false, pf, false, bvf[c],
                                                      (short)0, acc[c], false, false);
  }

  // write O / l in (b, n, dk, h) layout, f16, for the out-projection GEMM
#pragma unroll
  for (int c = 0; c < 4; ++c) {
    int dk = c * 16 + m;
#pragma unroll
    for (int r = 0; r < 8; ++r) {
      int qrow = qrow0 + rbase + r;
      float v = acc[c][r] / lrow[r];
      attnA[(size_t)(b * N_ + qrow) * DM + dk * HEADS_ + h] = (h16)v;
    }
  }
}

// ------------------------- residual + layernorm ----------------------------
// out = LN(x + y) * g + b ; optionally also writes an f16 copy.

template <bool WF32, bool WF16>
__global__ __launch_bounds__(256) void k_addln(
    const float* __restrict__ x, const float* __restrict__ y,
    const float* __restrict__ g, const float* __restrict__ bta,
    float* __restrict__ of, h16* __restrict__ oh) {
  __shared__ float red[256];
  const int row = blockIdx.x, t = threadIdx.x;
  const float* xr = x + (size_t)row * DM;
  const float* yr = y + (size_t)row * DM;
  float v0 = xr[t] + yr[t];
  float v1 = xr[t + 256] + yr[t + 256];
  red[t] = v0 + v1;
  __syncthreads();
  for (int s = 128; s > 0; s >>= 1) { if (t < s) red[t] += red[t + s]; __syncthreads(); }
  float mu = red[0] * (1.0f / DM);
  __syncthreads();
  float d0 = v0 - mu, d1 = v1 - mu;
  red[t] = d0 * d0 + d1 * d1;
  __syncthreads();
  for (int s = 128; s > 0; s >>= 1) { if (t < s) red[t] += red[t + s]; __syncthreads(); }
  float rstd = rsqrtf(red[0] * (1.0f / DM) + 1e-5f);
  float o0 = d0 * rstd * g[t] + bta[t];
  float o1 = d1 * rstd * g[t + 256] + bta[t + 256];
  if constexpr (WF32) { of[(size_t)row * DM + t] = o0; of[(size_t)row * DM + t + 256] = o1; }
  if constexpr (WF16) { oh[(size_t)row * DM + t] = (h16)o0; oh[(size_t)row * DM + t + 256] = (h16)o1; }
}

// ------------------------- host orchestration ------------------------------

extern "C" void kernel_launch(void* const* d_in, const int* in_sizes, int n_in,
                              void* d_out, int out_size, void* d_ws, size_t ws_size,
                              hipStream_t stream) {
  const float* q        = (const float*)d_in[0];
  const float* k        = (const float*)d_in[1];
  const float* v        = (const float*)d_in[2];
  const float* coords   = (const float*)d_in[3];
  /* d_in[4] = mask (all true) */
  const float* q_proj   = (const float*)d_in[5];
  const float* k_proj   = (const float*)d_in[6];
  const float* v_proj   = (const float*)d_in[7];
  const float* q_bias   = (const float*)d_in[8];
  const float* k_bias   = (const float*)d_in[9];
  const float* v_bias   = (const float*)d_in[10];
  const float* out_w    = (const float*)d_in[11];
  const float* spread_w = (const float*)d_in[12];
  const float* beta_w   = (const float*)d_in[13];
  const float* ln1_g    = (const float*)d_in[14];
  const float* ln1_b    = (const float*)d_in[15];
  const float* ln2_g    = (const float*)d_in[16];
  const float* ln2_b    = (const float*)d_in[17];
  const float* w1       = (const float*)d_in[18];
  const float* b1       = (const float*)d_in[19];
  const float* w2       = (const float*)d_in[20];
  const float* b2       = (const float*)d_in[21];

  char* ws = (char*)d_ws;
  size_t off = 0;
  auto alloc = [&](size_t bytes) -> char* {
    char* p = ws + off;
    off += (bytes + 255) & ~(size_t)255;
    return p;
  };
  h16* Wq   = (h16*)alloc((size_t)DM * DM * 2);
  h16* Wk   = (h16*)alloc((size_t)DM * DM * 2);
  h16* Wv   = (h16*)alloc((size_t)DM * DM * 2);
  h16* Wo   = (h16*)alloc((size_t)DM * DM * 2);
  h16* W1f  = (h16*)alloc((size_t)DM * DH * 2);
  h16* W2f  = (h16*)alloc((size_t)DH * DM * 2);
  h16* qf   = (h16*)alloc((size_t)ROWS * DM * 2);
  h16* kf   = (h16*)alloc((size_t)ROWS * DM * 2);   // kf+vf contiguous (4MB each)
  h16* vf   = (h16*)alloc((size_t)ROWS * DM * 2);
  h16* Qh   = (h16*)alloc((size_t)ROWS * DM * 2);   // Qh+Kh contiguous
  h16* Kh   = (h16*)alloc((size_t)ROWS * DM * 2);
  h16* Vh   = (h16*)alloc((size_t)ROWS * DM * 2);
  float* x1f = (float*)alloc((size_t)ROWS * DM * 4);
  h16* hbuf  = (h16*)alloc((size_t)ROWS * DH * 2);
  // lifetime-based aliases (inputs to a stage are dead after it runs):
  h16*   attnA   = qf;            // 4MB, after QKV proj qf is dead
  float* projtmp = (float*)kf;    // 8MB over kf+vf, dead after QKV proj
  float* ffnb    = (float*)Qh;    // 8MB over Qh+Kh, dead after attention+outproj
  h16*   x1h     = (h16*)Vh;      // 4MB, Vh dead after attention

  // 1) precision conversion / weight re-layout
  k_cvt_headw<<<(DM * DM) / 256, 256, 0, stream>>>(q_proj, Wq);
  k_cvt_headw<<<(DM * DM) / 256, 256, 0, stream>>>(k_proj, Wk);
  k_cvt_headw<<<(DM * DM) / 256, 256, 0, stream>>>(v_proj, Wv);
  k_cvt_transpose<<<(DM * DM) / 256, 256, 0, stream>>>(out_w, Wo, DM, DM);
  k_cvt_plain<<<(DM * DH) / 256, 256, 0, stream>>>(w1, W1f, DM * DH);
  k_cvt_plain<<<(DH * DM) / 256, 256, 0, stream>>>(w2, W2f, DH * DM);
  k_cvt_plain<<<(ROWS * DM) / 256, 256, 0, stream>>>(q, qf, ROWS * DM);
  k_cvt_plain<<<(ROWS * DM) / 256, 256, 0, stream>>>(k, kf, ROWS * DM);
  k_cvt_plain<<<(ROWS * DM) / 256, 256, 0, stream>>>(v, vf, ROWS * DM);

  // 2) QKV projections (+bias), f16 outputs in [b,n,h,dk] layout
  dim3 gq(DM / 64, ROWS / 64);
  k_gemm<true, false, false, true><<<gq, 128, 0, stream>>>(qf, Wq, q_bias, nullptr, Qh, ROWS, DM, DM);
  k_gemm<true, false, false, true><<<gq, 128, 0, stream>>>(kf, Wk, k_bias, nullptr, Kh, ROWS, DM, DM);
  k_gemm<true, false, false, true><<<gq, 128, 0, stream>>>(vf, Wv, v_bias, nullptr, Vh, ROWS, DM, DM);

  // 3) flash geometric attention -> attnA in (b,n,dk,h) layout
  dim3 ga(N_ / 64, HEADS_, B_);
  k_attn<<<ga, 128, 0, stream>>>(Qh, Kh, Vh, coords, spread_w, beta_w, attnA);

  // 4) output projection: attnA @ out_w.T
  k_gemm<false, false, true, false><<<gq, 128, 0, stream>>>(attnA, Wo, nullptr, projtmp, nullptr, ROWS, DM, DM);

  // 5) x1 = LN(q + proj)
  k_addln<true, true><<<ROWS, 256, 0, stream>>>(q, projtmp, ln1_g, ln1_b, x1f, x1h);

  // 6) FFN: relu(x1 @ w1 + b1) @ w2 + b2
  dim3 g1(DH / 64, ROWS / 64);
  k_gemm<true, true, false, true><<<g1, 128, 0, stream>>>(x1h, W1f, b1, nullptr, hbuf, ROWS, DH, DM);
  k_gemm<true, false, true, false><<<gq, 128, 0, stream>>>(hbuf, W2f, b2, ffnb, nullptr, ROWS, DM, DH);

  // 7) out = LN(x1 + ffn)
  k_addln<true, false><<<ROWS, 256, 0, stream>>>(x1f, ffnb, ln2_g, ln2_b, (float*)d_out, nullptr);

  (void)in_sizes; (void)n_in; (void)out_size; (void)ws_size;
}
